// GeneSymbolCNN_6923487281857
// MI455X (gfx1250) — matrix operand order repaired and checked by reading, not compiled, not run on hardware
//
#include <hip/hip_runtime.h>
#include <math.h>

typedef __attribute__((ext_vector_type(16))) _Float16 v16h;
typedef __attribute__((ext_vector_type(8)))  _Float16 v8h;
typedef __attribute__((ext_vector_type(8)))  float    v8f;

#define WAVES 16
#define SPB   16              // samples per block (one per wave)
#define EMB_ROW 40            // halfs per LDS emb row (32 + 8 pad) = 80B, conflict-friendly
#define EMB_ROWS 19           // 16 real rows + 3 zero pad rows (for shifted-A reads)
#define FEAT_STRIDE 104       // halfs (96 + 8 pad) = 208B, 16B aligned, conflict-friendly

__device__ __forceinline__ v16h cat16(v8h lo, v8h hi) {
  return __builtin_shufflevector(lo, hi, 0,1,2,3,4,5,6,7,8,9,10,11,12,13,14,15);
}

// A-matrix 16x32 f16 fragment (matches B): lanes0-15: K0-15;
// lanes16-31: K16-31. rowbase -> 32 contiguous halfs of this lane's row.
__device__ __forceinline__ v16h load_a_frag(const _Float16* rowbase, int grp) {
  v8h lo = *(const v8h*)(rowbase + grp * 16);
  v8h hi = *(const v8h*)(rowbase + grp * 16 + 8);
  return cat16(lo, hi);
}

// B-matrix 32x16 f16 fragment (contiguous K order, same as A): lane = column N; lanes0-15 hold K0-15,
// lanes16-31 hold K16-31, 2 K per VGPR. colbase -> 32 contiguous halfs (K0..31) of column.
__device__ __forceinline__ v16h load_b_frag(const _Float16* colbase, int grp) {
  v8h lo = *(const v8h*)(colbase + grp * 16);
  v8h hi = *(const v8h*)(colbase + grp * 16 + 8);
  return cat16(lo, hi);
}

__device__ __forceinline__ v8f wmma_f16(v16h a, v16h b, v8f c) {
  return __builtin_amdgcn_wmma_f32_16x16x32_f16(false, a, false, b, (short)0, c,
                                                false, false);
}

__global__ __launch_bounds__(512) void gene_cnn_kernel(
    const unsigned long long* __restrict__ x,   // [B,16] int64 tokens
    const float* __restrict__ embed,            // [67,32]
    const float* __restrict__ w2, const float* __restrict__ b2,   // [32,32,2],[32]
    const float* __restrict__ w3, const float* __restrict__ b3,   // [32,32,3],[32]
    const float* __restrict__ w4, const float* __restrict__ b4,   // [32,32,4],[32]
    const float* __restrict__ pw, const float* __restrict__ pb,   // [96,96],[96]
    float* __restrict__ out)                    // [B,96]
{
  __shared__ _Float16 sEmb[SPB * EMB_ROWS * EMB_ROW];  // 24320 B
  __shared__ _Float16 sW2[2 * 32 * 32];                // [s][c][e]
  __shared__ _Float16 sW3[3 * 32 * 32];
  __shared__ _Float16 sW4[4 * 32 * 32];
  __shared__ _Float16 sPW[96 * 96];                    // [o][f]
  __shared__ float    sBias[96];                       // b2|b3|b4
  __shared__ _Float16 sFeat[SPB * FEAT_STRIDE];        // [sample][feat]

  const int tid  = threadIdx.x;
  const int lane = tid & 31;
  const int wave = tid >> 5;
  const int col  = lane & 15;   // N-column / time-row / sample-row index
  const int grp  = lane >> 4;   // half-wave group (K-half selector)

  // ---- weights f32 -> f16 into LDS ([s][c][e] so a column's K run is contiguous) ----
  for (int i = tid; i < 2048; i += 512) {
    int s = i >> 10, c = (i >> 5) & 31, e = i & 31;
    sW2[i] = (_Float16)w2[c * 64 + e * 2 + s];
  }
  for (int i = tid; i < 3072; i += 512) {
    int s = i >> 10, c = (i >> 5) & 31, e = i & 31;
    sW3[i] = (_Float16)w3[c * 96 + e * 3 + s];
  }
  for (int i = tid; i < 4096; i += 512) {
    int s = i >> 10, c = (i >> 5) & 31, e = i & 31;
    sW4[i] = (_Float16)w4[c * 128 + e * 4 + s];
  }
  for (int i = tid; i < 9216; i += 512) sPW[i] = (_Float16)pw[i];
  if (tid < 96) sBias[tid] = (tid < 32) ? b2[tid] : (tid < 64 ? b3[tid - 32] : b4[tid - 64]);

  // ---- embedding gather: one sample per wave -> LDS f16 [19 rows x 32] ----
  const int samp = blockIdx.x * SPB + wave;
  _Float16* my = sEmb + wave * (EMB_ROWS * EMB_ROW);
  {
    const int r = col;      // time row 0..15
    const int part = grp;   // which 16 of the 32 embed dims
    unsigned tok = (unsigned)x[(size_t)samp * 16 + r];
    const float* src = embed + (size_t)tok * 32 + part * 16;
    v8f f0 = *(const v8f*)(src);
    v8f f1 = *(const v8f*)(src + 8);
    float msk = (tok == 0u) ? 0.0f : 1.0f;   // padding_idx=0 -> zero row
    f0 *= msk; f1 *= msk;
    v8h h0 = __builtin_convertvector(f0, v8h);
    v8h h1 = __builtin_convertvector(f1, v8h);
    _Float16* dst = my + r * EMB_ROW + part * 16;
    *(v8h*)(dst)     = h0;
    *(v8h*)(dst + 8) = h1;
    if (lane < 6) {  // zero pad rows 16..18 (fed to shifted-A tail rows; masked at pool)
      v8h zz = {};
      _Float16* z = my + (16 + (lane >> 1)) * EMB_ROW + (lane & 1) * 16;
      *(v8h*)(z) = zz; *(v8h*)(z + 8) = zz;
    }
  }
  __syncthreads();

  // ---- stage 1: conv-as-GEMM via shifted A fragments ----
  v8f a2_0 = {}, a2_1 = {}, a3_0 = {}, a3_1 = {}, a4_0 = {}, a4_1 = {};
#pragma unroll
  for (int s = 0; s < 4; ++s) {
    v16h A = load_a_frag(my + (col + s) * EMB_ROW, grp);  // rows t+s, K = emb dim
    {
      v16h B0 = load_b_frag(sW4 + s * 1024 + col * 32, grp);
      a4_0 = wmma_f16(A, B0, a4_0);
      v16h B1 = load_b_frag(sW4 + s * 1024 + (col + 16) * 32, grp);
      a4_1 = wmma_f16(A, B1, a4_1);
    }
    if (s < 3) {
      v16h B0 = load_b_frag(sW3 + s * 1024 + col * 32, grp);
      a3_0 = wmma_f16(A, B0, a3_0);
      v16h B1 = load_b_frag(sW3 + s * 1024 + (col + 16) * 32, grp);
      a3_1 = wmma_f16(A, B1, a3_1);
    }
    if (s < 2) {
      v16h B0 = load_b_frag(sW2 + s * 1024 + col * 32, grp);
      a2_0 = wmma_f16(A, B0, a2_0);
      v16h B1 = load_b_frag(sW2 + s * 1024 + (col + 16) * 32, grp);
      a2_1 = wmma_f16(A, B1, a2_1);
    }
  }

  // ---- max over time (+bias, ReLU) exploiting the C layout:
  // lanes0-15 hold rows 0-7 (always valid), lanes16-31 rows 8-15 (tail masked) ----
  auto pool = [&](v8f acc, int T, int cAbs) {
    const int nv = grp ? (T - 8) : 8;  // #valid accumulator rows for this half-wave
    float m = -3.402823466e38f;
#pragma unroll
    for (int v = 0; v < 8; ++v) m = (v < nv) ? fmaxf(m, acc[v]) : m;
    m = fmaxf(m, __shfl_xor(m, 16, 32));          // merge row halves (same channel)
    m = fmaxf(m + sBias[cAbs], 0.0f);             // relu(max_t + b) == max_t relu(.+b)
    if (lane < 16) sFeat[wave * FEAT_STRIDE + cAbs] = (_Float16)m;
  };
  pool(a2_0, 15, col);      pool(a2_1, 15, col + 16);   // g2 -> feat[0:32)
  pool(a3_0, 14, 32 + col); pool(a3_1, 14, 48 + col);   // g3 -> feat[32:64)
  pool(a4_0, 13, 64 + col); pool(a4_1, 13, 80 + col);   // g4 -> feat[64:96)
  __syncthreads();

  // ---- stage 2: [16 samples x 96] @ pw^T, waves 0..5 take one 16-ch output tile ----
  if (wave < 6) {
    const int n = wave;
    const int o = n * 16 + col;
    v8f acc = {};
#pragma unroll
    for (int q = 0; q < 3; ++q) {
      const _Float16* ar = sFeat + col * FEAT_STRIDE + q * 32;  // A row = sample `col`
      v16h A = cat16(*(const v8h*)(ar + grp * 16),
                     *(const v8h*)(ar + grp * 16 + 8));
      const _Float16* br = sPW + o * 96 + q * 32;               // B column = output o
      v16h Bf = cat16(*(const v8h*)(br + grp * 16),
                      *(const v8h*)(br + grp * 16 + 8));
      acc = wmma_f16(A, Bf, acc);
    }
    const float pbv = pb[o];
    const size_t obase = (size_t)blockIdx.x * SPB * 96;
#pragma unroll
    for (int v = 0; v < 8; ++v) {
      int m = v + 8 * grp;  // sample row within block
      out[obase + (size_t)m * 96 + o] = fmaxf(acc[v] + pbv, 0.0f);
    }
  }
}

extern "C" void kernel_launch(void* const* d_in, const int* in_sizes, int n_in,
                              void* d_out, int out_size, void* d_ws, size_t ws_size,
                              hipStream_t stream) {
  const unsigned long long* x = (const unsigned long long*)d_in[0];  // int64 tokens
  const float* embed = (const float*)d_in[1];
  const float* w2 = (const float*)d_in[2];
  const float* b2 = (const float*)d_in[3];
  const float* w3 = (const float*)d_in[4];
  const float* b3 = (const float*)d_in[5];
  const float* w4 = (const float*)d_in[6];
  const float* b4 = (const float*)d_in[7];
  const float* pw = (const float*)d_in[8];
  const float* pb = (const float*)d_in[9];
  float* out = (float*)d_out;

  const int nSamples = in_sizes[0] / 16;   // B = 131072
  const int blocks = nSamples / SPB;       // 8192 blocks of 16 waves
  hipLaunchKernelGGL(gene_cnn_kernel, dim3(blocks), dim3(512), 0, stream,
                     x, embed, w2, b2, w3, b3, w4, b4, pw, pb, out);
}
